// GraphSAGE_40063454937586
// MI455X (gfx1250) — matrix operand order repaired and checked by reading, not compiled, not run on hardware
//
#include <hip/hip_runtime.h>

#define N_NODES 50000
#define E_EDGES 800000
#define F_IN    128
#define M_TILES ((N_NODES + 15) / 16)   // 3125
#define NPAD    50016                   // N rounded up, keeps 32B ws alignment

typedef __attribute__((ext_vector_type(16))) _Float16 v16h;
typedef __attribute__((ext_vector_type(8)))  _Float16 v8h;
typedef __attribute__((ext_vector_type(4)))  _Float16 v4h;
typedef __attribute__((ext_vector_type(8)))  float    v8f;

// ---------------- utility ----------------
__global__ void zero_u32(unsigned* __restrict__ p, int n) {
    int i = blockIdx.x * blockDim.x + threadIdx.x;
    if (i < n) p[i] = 0u;
}

// f32 -> f16, vectorized x4
__global__ void f32_to_f16(const float* __restrict__ a, _Float16* __restrict__ b, int n4) {
    int i = blockIdx.x * blockDim.x + threadIdx.x;
    if (i < n4) {
        float4 v = *(const float4*)(a + (long long)i * 4);
        v4h o;
        o[0] = (_Float16)v.x; o[1] = (_Float16)v.y;
        o[2] = (_Float16)v.z; o[3] = (_Float16)v.w;
        *(v4h*)(b + (long long)i * 4) = o;
    }
}

// ---------------- degree counting ----------------
__global__ void count_edges(const int* __restrict__ dst, unsigned* __restrict__ cnt, int e) {
    int i = blockIdx.x * blockDim.x + threadIdx.x;
    if (i < e) {
        __hip_atomic_fetch_add(&cnt[dst[i]], 1u, __ATOMIC_RELAXED, __HIP_MEMORY_SCOPE_AGENT);
    }
}

__global__ void make_recip(const unsigned* __restrict__ cnt, float* __restrict__ recip, int n) {
    int i = blockIdx.x * blockDim.x + threadIdx.x;
    if (i < n) {
        unsigned c = cnt[i];
        recip[i] = 1.0f / (float)(c ? c : 1u);
    }
}

// ---------------- hierarchical exclusive scan (CSR row offsets) ----------------
__global__ void scan_block(const unsigned* __restrict__ cnt, int* __restrict__ excl,
                           int* __restrict__ bsum, int n) {
    __shared__ int s[256];
    int i = blockIdx.x * 256 + threadIdx.x;
    int v = (i < n) ? (int)cnt[i] : 0;
    s[threadIdx.x] = v;
    __syncthreads();
#pragma unroll
    for (int off = 1; off < 256; off <<= 1) {
        int t = (threadIdx.x >= off) ? s[threadIdx.x - off] : 0;
        __syncthreads();
        s[threadIdx.x] += t;
        __syncthreads();
    }
    if (i < n) excl[i] = s[threadIdx.x] - v;
    if (threadIdx.x == 255) bsum[blockIdx.x] = s[255];
}

__global__ void scan_bsum(int* __restrict__ bsum, int nb) {   // nb <= 256, one block
    __shared__ int s[256];
    int v = (threadIdx.x < nb) ? bsum[threadIdx.x] : 0;
    s[threadIdx.x] = v;
    __syncthreads();
#pragma unroll
    for (int off = 1; off < 256; off <<= 1) {
        int t = (threadIdx.x >= off) ? s[threadIdx.x - off] : 0;
        __syncthreads();
        s[threadIdx.x] += t;
        __syncthreads();
    }
    if (threadIdx.x < nb) bsum[threadIdx.x] = s[threadIdx.x] - v;
}

__global__ void scan_add(int* __restrict__ excl, const int* __restrict__ bsum, int n) {
    int i = blockIdx.x * 256 + threadIdx.x;
    if (i < n) excl[i] += bsum[blockIdx.x];
    if (blockIdx.x == 0 && threadIdx.x == 0) excl[n] = E_EDGES;   // row_start[N]
}

__global__ void copy_i32(const int* __restrict__ a, int* __restrict__ b, int n) {
    int i = blockIdx.x * blockDim.x + threadIdx.x;
    if (i < n) b[i] = a[i];
}

// ---------------- CSR fill: 800K int atomics, done once ----------------
__global__ void fill_eidx(const int* __restrict__ src, const int* __restrict__ dst,
                          int* __restrict__ cursor, int* __restrict__ eidx, int e) {
    int i = blockIdx.x * blockDim.x + threadIdx.x;
    if (i < e) {
        int d = dst[i];
        int pos = __hip_atomic_fetch_add(&cursor[d], 1, __ATOMIC_RELAXED, __HIP_MEMORY_SCOPE_AGENT);
        eidx[pos] = src[i];
    }
}

// ---------------- gather mean (f16 rows, f32 accumulation) ----------------
// One wave per node; lane L owns features [4L,4L+4) (8B b64 per lane, 256B
// coalesced per row). Unrolled x4 so 4 independent row loads are in flight.
__global__ void gather_mean16(const _Float16* __restrict__ h,
                              const int* __restrict__ row_start,
                              const int* __restrict__ eidx,
                              const float* __restrict__ recip,
                              _Float16* __restrict__ mean) {
    int wave = threadIdx.x >> 5;
    int lane = threadIdx.x & 31;
    int node = blockIdx.x * (blockDim.x >> 5) + wave;
    if (node >= N_NODES) return;
    int beg = row_start[node];
    int end = row_start[node + 1];
    float ax = 0.f, ay = 0.f, az = 0.f, aw = 0.f;
    int j = beg;
    for (; j + 4 <= end; j += 4) {
        int s0 = eidx[j], s1 = eidx[j + 1], s2 = eidx[j + 2], s3 = eidx[j + 3];
        v4h v0 = *(const v4h*)(h + (long long)s0 * F_IN + lane * 4);
        v4h v1 = *(const v4h*)(h + (long long)s1 * F_IN + lane * 4);
        v4h v2 = *(const v4h*)(h + (long long)s2 * F_IN + lane * 4);
        v4h v3 = *(const v4h*)(h + (long long)s3 * F_IN + lane * 4);
        ax += (float)v0[0] + (float)v1[0] + (float)v2[0] + (float)v3[0];
        ay += (float)v0[1] + (float)v1[1] + (float)v2[1] + (float)v3[1];
        az += (float)v0[2] + (float)v1[2] + (float)v2[2] + (float)v3[2];
        aw += (float)v0[3] + (float)v1[3] + (float)v2[3] + (float)v3[3];
    }
    for (; j < end; ++j) {
        int s = eidx[j];
        v4h v = *(const v4h*)(h + (long long)s * F_IN + lane * 4);
        ax += (float)v[0]; ay += (float)v[1]; az += (float)v[2]; aw += (float)v[3];
    }
    float r = recip[node];
    v4h o;
    o[0] = (_Float16)(ax * r); o[1] = (_Float16)(ay * r);
    o[2] = (_Float16)(az * r); o[3] = (_Float16)(aw * r);
    *(v4h*)(mean + (long long)node * F_IN + lane * 4) = o;
}

// ---------------- weight fragment pre-pack (f32 [F_IN x Fo] -> f16 WMMA B frags) ----
__global__ void pack_w(const float* __restrict__ W, _Float16* __restrict__ P, int Fo) {
    int t = blockIdx.x * blockDim.x + threadIdx.x;
    int total = (Fo / 16) * (F_IN / 32) * 32;
    if (t >= total) return;
    int lane  = t & 31;
    int kkIdx = (t >> 5) & ((F_IN / 32) - 1);
    int ntile = t >> 7;
    int col = ntile * 16 + (lane & 15);
    int kB  = kkIdx * 32 + (lane >> 4) * 8;
    _Float16* o = P + (size_t)t * 16;
#pragma unroll
    for (int i = 0; i < 16; ++i)
        o[i] = (_Float16)W[(long long)(kB + (i < 8 ? i : i + 8)) * Fo + col];
}

// ---------------- fused dual-GEMM SAGE layer via WMMA ----------------
// out[m,n] = relu?( mean16[m,:] @ Wl + bl + h16[m,:] @ Wr )
// A-operands are pure 16B f16 loads + register concat (no conversion VALU);
// B fragments staged through 8KB LDS. f32 accumulate; store f16 (intermediate)
// or f32 (final output).
__global__ void sage_gemm(const _Float16* __restrict__ mean,
                          const _Float16* __restrict__ hroot,
                          const _Float16* __restrict__ WlP,
                          const float* __restrict__ bl,
                          const _Float16* __restrict__ WrP,
                          float* __restrict__ outF,       // used if write16 == 0
                          _Float16* __restrict__ outH,    // used if write16 == 1
                          int Fo, int do_relu, int write16) {
    __shared__ v16h sB[2][F_IN / 32][32];     // 8KB

    const int tid  = threadIdx.x;             // 0..127
    const int lane = tid & 31;
    const int wave = tid >> 5;
    const int half = lane >> 4;

    {
        const size_t base = (size_t)blockIdx.y * (F_IN / 32) * 32;
        sB[0][tid >> 5][tid & 31] = *((const v16h*)WlP + base + tid);
        sB[1][tid >> 5][tid & 31] = *((const v16h*)WrP + base + tid);
    }
    __syncthreads();

    const int mtile = blockIdx.x * 4 + wave;  // wave-uniform ragged guard
    if (mtile >= M_TILES) return;

    const int m0   = mtile * 16;
    const int mrow = m0 + (lane & 15);
    const int col  = blockIdx.y * 16 + (lane & 15);

    const float bias = bl[col];
    v8f c;
#pragma unroll
    for (int j = 0; j < 8; ++j) c[j] = bias;

    const _Float16* __restrict__ pa = mean  + (long long)mrow * F_IN;
    const _Float16* __restrict__ ph = hroot + (long long)mrow * F_IN;

#pragma unroll
    for (int kkIdx = 0; kkIdx < F_IN / 32; ++kkIdx) {
        const int kA = kkIdx * 32 + half * 8;
        v8h al = *(const v8h*)(pa + kA);
        v8h ah = *(const v8h*)(pa + kA + 16);
        v8h rl = *(const v8h*)(ph + kA);
        v8h rh = *(const v8h*)(ph + kA + 16);
        v16h a  = __builtin_shufflevector(al, ah, 0, 1, 2, 3, 4, 5, 6, 7,
                                                  8, 9, 10, 11, 12, 13, 14, 15);
        v16h ar = __builtin_shufflevector(rl, rh, 0, 1, 2, 3, 4, 5, 6, 7,
                                                  8, 9, 10, 11, 12, 13, 14, 15);
        v16h b  = sB[0][kkIdx][lane];
        v16h br = sB[1][kkIdx][lane];

        c = __builtin_amdgcn_wmma_f32_16x16x32_f16(false, a,  false, b,  (short)0, c, false, false);
        c = __builtin_amdgcn_wmma_f32_16x16x32_f16(false, ar, false, br, (short)0, c, false, false);
    }

    if (do_relu) {
#pragma unroll
        for (int j = 0; j < 8; ++j) c[j] = fmaxf(c[j], 0.0f);
    }

    if (write16) {
#pragma unroll
        for (int j = 0; j < 8; ++j)
            outH[(long long)(m0 + half * 8 + j) * Fo + col] = (_Float16)c[j];
    } else {
#pragma unroll
        for (int j = 0; j < 8; ++j)
            outF[(long long)(m0 + half * 8 + j) * Fo + col] = c[j];
    }
}

// ---------------- driver ----------------
extern "C" void kernel_launch(void* const* d_in, const int* in_sizes, int n_in,
                              void* d_out, int out_size, void* d_ws, size_t ws_size,
                              hipStream_t stream) {
    const float* x   = (const float*)d_in[0];
    const int*   ei  = (const int*)d_in[1];
    const int*   src = ei;
    const int*   dst = ei + E_EDGES;
    const float* Wl0 = (const float*)d_in[2];
    const float* bl0 = (const float*)d_in[3];
    const float* Wr0 = (const float*)d_in[4];
    const float* Wl1 = (const float*)d_in[5];
    const float* bl1 = (const float*)d_in[6];
    const float* Wr1 = (const float*)d_in[7];
    const float* Wl2 = (const float*)d_in[8];
    const float* bl2 = (const float*)d_in[9];
    const float* Wr2 = (const float*)d_in[10];

    // workspace layout (every chunk a multiple of 32B)
    const size_t FEAT = (size_t)N_NODES * F_IN;                  // 6.4M elems
    _Float16* mean16 = (_Float16*)d_ws;                          // FEAT f16
    _Float16* x16    = mean16 + FEAT;                            // FEAT f16
    _Float16* h1     = x16    + FEAT;                            // FEAT f16
    _Float16* h2     = h1     + FEAT;                            // FEAT f16
    float*    recip  = (float*)(h2 + FEAT);                      // NPAD f32
    unsigned* cnt    = (unsigned*)(recip + NPAD);                // NPAD u32
    int*      rowst  = (int*)(cnt + NPAD);                       // NPAD
    int*      curs   = rowst + NPAD;                             // NPAD
    int*      bsum   = curs + NPAD;                              // 256
    int*      eidx   = bsum + 256;                               // E ints
    _Float16* pWl0   = (_Float16*)(eidx + E_EDGES);
    _Float16* pWr0   = pWl0 + 8 * 4 * 32 * 16;
    _Float16* pWl1   = pWr0 + 8 * 4 * 32 * 16;
    _Float16* pWr1   = pWl1 + 8 * 4 * 32 * 16;
    _Float16* pWl2   = pWr1 + 8 * 4 * 32 * 16;
    _Float16* pWr2   = pWl2 + 4 * 4 * 32 * 16;

    const int ngrid  = (N_NODES + 255) / 256;                    // 196
    const int egrid  = (E_EDGES + 255) / 256;
    const int ggrid  = (M_TILES + 3) / 4;                        // 782
    const int gagrid = (N_NODES + 7) / 8;                        // 8 waves/block
    const int cvt4   = (int)(FEAT / 4);
    const int cgrid  = (cvt4 + 255) / 256;

    // ---- build CSR + recip (fresh every call) ----
    zero_u32   <<<ngrid, 256, 0, stream>>>(cnt, N_NODES);
    count_edges<<<egrid, 256, 0, stream>>>(dst, cnt, E_EDGES);
    make_recip <<<ngrid, 256, 0, stream>>>(cnt, recip, N_NODES);
    scan_block <<<ngrid, 256, 0, stream>>>(cnt, rowst, bsum, N_NODES);
    scan_bsum  <<<1,     256, 0, stream>>>(bsum, ngrid);
    scan_add   <<<ngrid, 256, 0, stream>>>(rowst, bsum, N_NODES);
    copy_i32   <<<ngrid, 256, 0, stream>>>(rowst, curs, N_NODES);
    fill_eidx  <<<egrid, 256, 0, stream>>>(src, dst, curs, eidx, E_EDGES);

    // ---- one-time conversions ----
    f32_to_f16<<<cgrid, 256, 0, stream>>>(x, x16, cvt4);
    pack_w<<<4, 256, 0, stream>>>(Wl0, pWl0, 128);
    pack_w<<<4, 256, 0, stream>>>(Wr0, pWr0, 128);
    pack_w<<<4, 256, 0, stream>>>(Wl1, pWl1, 128);
    pack_w<<<4, 256, 0, stream>>>(Wr1, pWr1, 128);
    pack_w<<<2, 256, 0, stream>>>(Wl2, pWl2, 64);
    pack_w<<<2, 256, 0, stream>>>(Wr2, pWr2, 64);

    // ---- layer 0: x -> h1 (relu, f16 out) ----
    gather_mean16<<<gagrid, 256, 0, stream>>>(x16, rowst, eidx, recip, mean16);
    sage_gemm<<<dim3(ggrid, 128 / 16), 128, 0, stream>>>(
        mean16, x16, pWl0, bl0, pWr0, nullptr, h1, 128, 1, 1);

    // ---- layer 1: h1 -> h2 (relu, f16 out) ----
    gather_mean16<<<gagrid, 256, 0, stream>>>(h1, rowst, eidx, recip, mean16);
    sage_gemm<<<dim3(ggrid, 128 / 16), 128, 0, stream>>>(
        mean16, h1, pWl1, bl1, pWr1, nullptr, h2, 128, 1, 1);

    // ---- layer 2: h2 -> out (no relu, f32 out) ----
    gather_mean16<<<gagrid, 256, 0, stream>>>(h2, rowst, eidx, recip, mean16);
    sage_gemm<<<dim3(ggrid, 64 / 16), 128, 0, stream>>>(
        mean16, h2, pWl2, bl2, pWr2, (float*)d_out, nullptr, 64, 0, 0);
}
